// Attention_83056077570148
// MI455X (gfx1250) — matrix operand order repaired and checked by reading, not compiled, not run on hardware
//
#include <hip/hip_runtime.h>
#include <hip/hip_bf16.h>
#include <stdint.h>

// ---------------------------------------------------------------------------
// Types / WMMA helper
// ---------------------------------------------------------------------------
typedef _Float16 half16 __attribute__((ext_vector_type(16)));
typedef _Float16 half8  __attribute__((ext_vector_type(8)));
typedef float    float8 __attribute__((ext_vector_type(8)));

union HF16 { half16 v; half8 h[2]; };

__device__ __forceinline__ float8 wmma16(const HF16& a, const HF16& b, float8 c) {
  // v_wmma_f32_16x16x32_f16  (D = A x B + C), wave32
  return __builtin_amdgcn_wmma_f32_16x16x32_f16(false, a.v, false, b.v,
                                                (short)0, c, false, false);
}

#define BATCH 4
#define SEQ   1024
#define CH    1024
#define HEADS 16
#define HDIM  64

// ---------------------------------------------------------------------------
// Kernel 0: f32 -> f16 conversion
// ---------------------------------------------------------------------------
__global__ void cvt_f32_to_f16(const float* __restrict__ in,
                               _Float16* __restrict__ out, int n) {
  int i = blockIdx.x * blockDim.x + threadIdx.x;
  if (i < n) out[i] = (_Float16)in[i];
}

// ---------------------------------------------------------------------------
// Kernel 1: Out[M,Nn] = A[M,K](f16) * W[Nn,K](f16)^T + bias (f32 out)
// 8 waves/WG; each wave computes a 32x64 tile (two 16-row A tiles sharing
// four 16-col B tiles -> 8 WMMAs per K-step, 12 b128 loads).
// Double-buffered K loop: next iteration's fragments are issued before the
// current WMMAs so the s_wait before the WMMAs is partial, overlapping
// global loads with matrix ops.
// A-frag: interleaved K runs of 8 (ISA A 16x32 layout).
// B-frag: interleaved K runs of 8, same order as A-frag.
// ---------------------------------------------------------------------------
__global__ void wmma_gemm_bias(const _Float16* __restrict__ A,
                               const _Float16* __restrict__ W,
                               const float* __restrict__ bias,
                               float* __restrict__ Out,
                               int M, int Nn, int K) {
  const int wave  = threadIdx.x >> 5;
  const int lane  = threadIdx.x & 31;
  const int lhalf = lane & 15;
  const int koffA = (lane < 16) ? 0 : 8;    // A interleave
  const int koffB = (lane < 16) ? 0 : 8;    // B interleave (matches A)

  const int rowbase = blockIdx.x * 256 + wave * 32;
  const int colbase = blockIdx.y * 64;

  const _Float16* arow0 = A + (size_t)(rowbase + lhalf) * K;
  const _Float16* arow1 = A + (size_t)(rowbase + 16 + lhalf) * K;
  const _Float16* wrow[4];
#pragma unroll
  for (int t = 0; t < 4; ++t)
    wrow[t] = W + (size_t)(colbase + t * 16 + lhalf) * K;

  float8 acc0[4] = {};
  float8 acc1[4] = {};

  // preload K-step 0
  HF16 a0c, a1c, bc[4];
  a0c.h[0] = *(const half8*)(arow0 + koffA);
  a0c.h[1] = *(const half8*)(arow0 + 16 + koffA);
  a1c.h[0] = *(const half8*)(arow1 + koffA);
  a1c.h[1] = *(const half8*)(arow1 + 16 + koffA);
#pragma unroll
  for (int t = 0; t < 4; ++t) {
    bc[t].h[0] = *(const half8*)(wrow[t] + koffB);
    bc[t].h[1] = *(const half8*)(wrow[t] + koffB + 16);
  }

  for (int k = 0; k < K; k += 32) {
    // issue next iteration's loads first (last iter redundantly reloads k)
    const int kn = (k + 32 < K) ? (k + 32) : k;
    HF16 a0n, a1n, bn[4];
    a0n.h[0] = *(const half8*)(arow0 + kn + koffA);
    a0n.h[1] = *(const half8*)(arow0 + kn + 16 + koffA);
    a1n.h[0] = *(const half8*)(arow1 + kn + koffA);
    a1n.h[1] = *(const half8*)(arow1 + kn + 16 + koffA);
#pragma unroll
    for (int t = 0; t < 4; ++t) {
      bn[t].h[0] = *(const half8*)(wrow[t] + kn + koffB);
      bn[t].h[1] = *(const half8*)(wrow[t] + kn + koffB + 16);
    }

    // compute on current fragments (B reused across both row tiles)
#pragma unroll
    for (int t = 0; t < 4; ++t) acc0[t] = wmma16(a0c, bc[t], acc0[t]);
#pragma unroll
    for (int t = 0; t < 4; ++t) acc1[t] = wmma16(a1c, bc[t], acc1[t]);

    a0c = a0n;
    a1c = a1n;
#pragma unroll
    for (int t = 0; t < 4; ++t) bc[t] = bn[t];
  }

#pragma unroll
  for (int t = 0; t < 4; ++t) {
    const int col = colbase + t * 16 + lhalf;
    const float bv = bias[col];
#pragma unroll
    for (int r = 0; r < 8; ++r) {
      const int rowm = r + ((lane >= 16) ? 8 : 0);
      Out[(size_t)(rowbase + rowm) * Nn + col]      = acc0[t][r] + bv;
      Out[(size_t)(rowbase + 16 + rowm) * Nn + col] = acc1[t][r] + bv;
    }
  }
}

// ---------------------------------------------------------------------------
// Kernel 2: per-row LayerNorm of q and k (over full C) + split to
//  qh[B,H,N,D] (f16, pre-scaled by D^-0.5), kh[B,H,N,D] (f16),
//  vT[B,H,D,N] (f16, transposed for contiguous B-fragments in attn*V).
// One 256-thread block per token row.
// ---------------------------------------------------------------------------
__global__ void qk_ln_split(const float* __restrict__ qkv,
                            const float* __restrict__ qn_w, const float* __restrict__ qn_b,
                            const float* __restrict__ kn_w, const float* __restrict__ kn_b,
                            _Float16* __restrict__ qh,
                            _Float16* __restrict__ kh,
                            _Float16* __restrict__ vT) {
  const int row = blockIdx.x;          // 0..B*N-1
  const int b = row >> 10;
  const int n = row & 1023;
  const int tid = threadIdx.x;

  __shared__ float s1[256];
  __shared__ float s2[256];

  const float* base = qkv + (size_t)row * (3 * CH);

#pragma unroll 1
  for (int part = 0; part < 2; ++part) {
    const float* src = base + part * CH;
    float vbuf[4];
    float sum = 0.f, sumsq = 0.f;
#pragma unroll
    for (int i = 0; i < 4; ++i) {
      const int c = tid + i * 256;
      const float x = src[c];
      vbuf[i] = x;
      sum += x;
      sumsq += x * x;
    }
    s1[tid] = sum;
    s2[tid] = sumsq;
    __syncthreads();
    for (int off = 128; off > 0; off >>= 1) {
      if (tid < off) { s1[tid] += s1[tid + off]; s2[tid] += s2[tid + off]; }
      __syncthreads();
    }
    const float mu  = s1[0] * (1.f / CH);
    const float var = s2[0] * (1.f / CH) - mu * mu;
    const float rs  = rsqrtf(var + 1e-5f);
    __syncthreads();   // allow s1/s2 reuse next part

    const float* w  = part ? kn_w : qn_w;
    const float* bb = part ? kn_b : qn_b;
    _Float16* dst   = part ? kh : qh;
    const float sc  = part ? 1.0f : 0.125f;   // D^-0.5 folded into q
#pragma unroll
    for (int i = 0; i < 4; ++i) {
      const int c = tid + i * 256;
      const float y = ((vbuf[i] - mu) * rs * w[c] + bb[c]) * sc;
      const int hh = c >> 6, d = c & 63;
      dst[(((size_t)(b * HEADS + hh) * SEQ) + n) * HDIM + d] = (_Float16)y;
    }
  }

  // v: plain convert, stored transposed [B,H,D,N]
  const float* vsrc = base + 2 * CH;
#pragma unroll
  for (int i = 0; i < 4; ++i) {
    const int c = tid + i * 256;
    const int hh = c >> 6, d = c & 63;
    vT[(((size_t)(b * HEADS + hh) * HDIM) + d) * SEQ + n] = (_Float16)vsrc[c];
  }
}

// ---------------------------------------------------------------------------
// Kernel 3: flash attention.  grid.x = B*H*(N/128); 8 waves/WG, each wave
// owns 16 query rows, online softmax over 64-key blocks.
// ---------------------------------------------------------------------------
__global__ void flash_attn(const _Float16* __restrict__ qh,
                           const _Float16* __restrict__ kh,
                           const _Float16* __restrict__ vT,
                           _Float16* __restrict__ attno) {
  const int wave  = threadIdx.x >> 5;
  const int lane  = threadIdx.x & 31;
  const int lhalf = lane & 15;
  const int koffA = (lane < 16) ? 0 : 8;
  const int koffB = (lane < 16) ? 0 : 8;

  const int bh     = blockIdx.x >> 3;        // 0..63
  const int rowblk = blockIdx.x & 7;
  const int b = bh >> 4, h = bh & 15;
  const int rowbase = rowblk * 128 + wave * 16;

  __shared__ __align__(16) _Float16 pbuf[8][16 * 64];   // per-wave P tile
  _Float16* myp = &pbuf[wave][0];

  const _Float16* qbase = qh + (size_t)bh * SEQ * HDIM;
  const _Float16* kbase = kh + (size_t)bh * SEQ * HDIM;
  const _Float16* vbase = vT + (size_t)bh * HDIM * SEQ;

  // load the wave's q rows once: two A fragments (d = 0..31, 32..63)
  const _Float16* qrow = qbase + (size_t)(rowbase + lhalf) * HDIM;
  HF16 qa0, qa1;
  qa0.h[0] = *(const half8*)(qrow + koffA);
  qa0.h[1] = *(const half8*)(qrow + 16 + koffA);
  qa1.h[0] = *(const half8*)(qrow + 32 + koffA);
  qa1.h[1] = *(const half8*)(qrow + 48 + koffA);

  float8 acco[4] = {};
  float m_run[8], l_run[8];
#pragma unroll
  for (int r = 0; r < 8; ++r) { m_run[r] = -1e30f; l_run[r] = 0.f; }

  for (int kb = 0; kb < SEQ / 64; ++kb) {
    const int keyb = kb * 64;

    // ---- S = q * k^T (16 x 64), 4 column tiles, 2 k-steps each ----
    float8 s[4];
#pragma unroll
    for (int t = 0; t < 4; ++t) {
      const _Float16* krow = kbase + (size_t)(keyb + t * 16 + lhalf) * HDIM;
      HF16 b0, b1;
      b0.h[0] = *(const half8*)(krow + koffB);
      b0.h[1] = *(const half8*)(krow + koffB + 16);
      b1.h[0] = *(const half8*)(krow + 32 + koffB);
      b1.h[1] = *(const half8*)(krow + 32 + koffB + 16);
      float8 z = {};
      z = wmma16(qa0, b0, z);
      s[t] = wmma16(qa1, b1, z);
    }

    // ---- online softmax update ----
    float mnew[8];
#pragma unroll
    for (int r = 0; r < 8; ++r) {
      float tm = fmaxf(fmaxf(s[0][r], s[1][r]), fmaxf(s[2][r], s[3][r]));
      tm = fmaxf(tm, __shfl_xor(tm, 1));
      tm = fmaxf(tm, __shfl_xor(tm, 2));
      tm = fmaxf(tm, __shfl_xor(tm, 4));
      tm = fmaxf(tm, __shfl_xor(tm, 8));
      mnew[r] = fmaxf(m_run[r], tm);
      const float sc = __expf(m_run[r] - mnew[r]);
      l_run[r] *= sc;
#pragma unroll
      for (int t = 0; t < 4; ++t) acco[t][r] *= sc;
    }

    // ---- P = exp(S - m), row sums, spill P (f16) to LDS ----
#pragma unroll
    for (int r = 0; r < 8; ++r) {
      const int rowm = r + ((lane >= 16) ? 8 : 0);
      float ps = 0.f;
#pragma unroll
      for (int t = 0; t < 4; ++t) {
        const float p = __expf(s[t][r] - mnew[r]);
        ps += p;
        myp[rowm * 64 + t * 16 + lhalf] = (_Float16)p;
      }
      ps += __shfl_xor(ps, 1);
      ps += __shfl_xor(ps, 2);
      ps += __shfl_xor(ps, 4);
      ps += __shfl_xor(ps, 8);
      l_run[r] += ps;
      m_run[r] = mnew[r];
    }
    __syncthreads();

    // ---- re-load P as A fragments (K = key dim, 2 k-steps) ----
    const _Float16* prow = myp + lhalf * 64;
    HF16 pa0, pa1;
    pa0.h[0] = *(const half8*)(prow + koffA);
    pa0.h[1] = *(const half8*)(prow + 16 + koffA);
    pa1.h[0] = *(const half8*)(prow + 32 + koffA);
    pa1.h[1] = *(const half8*)(prow + 48 + koffA);

    // ---- O += P * V : V^T stored [d][n] so B-frags are contiguous ----
#pragma unroll
    for (int t = 0; t < 4; ++t) {
      const _Float16* vcol = vbase + (size_t)(t * 16 + lhalf) * SEQ + keyb;
      HF16 v0, v1;
      v0.h[0] = *(const half8*)(vcol + koffB);
      v0.h[1] = *(const half8*)(vcol + koffB + 16);
      v1.h[0] = *(const half8*)(vcol + 32 + koffB);
      v1.h[1] = *(const half8*)(vcol + 32 + koffB + 16);
      acco[t] = wmma16(pa0, v0, acco[t]);
      acco[t] = wmma16(pa1, v1, acco[t]);
    }
    __syncthreads();
  }

  // ---- normalize + store attention output (f16, [B,N,C] layout) ----
#pragma unroll
  for (int t = 0; t < 4; ++t) {
    const int d = t * 16 + lhalf;
#pragma unroll
    for (int r = 0; r < 8; ++r) {
      const int rowm = r + ((lane >= 16) ? 8 : 0);
      const int n = rowbase + rowm;
      attno[((size_t)(b * SEQ + n)) * CH + h * HDIM + d] =
          (_Float16)(acco[t][r] / l_run[r]);
    }
  }
}

// ---------------------------------------------------------------------------
// Launch
// ---------------------------------------------------------------------------
extern "C" void kernel_launch(void* const* d_in, const int* in_sizes, int n_in,
                              void* d_out, int out_size, void* d_ws, size_t ws_size,
                              hipStream_t stream) {
  const float* x      = (const float*)d_in[0];
  const float* qkv_w  = (const float*)d_in[1];
  const float* qkv_b  = (const float*)d_in[2];
  const float* qn_w   = (const float*)d_in[3];
  const float* qn_b   = (const float*)d_in[4];
  const float* kn_w   = (const float*)d_in[5];
  const float* kn_b   = (const float*)d_in[6];
  const float* proj_w = (const float*)d_in[7];
  const float* proj_b = (const float*)d_in[8];

  const int M = BATCH * SEQ;      // 4096
  uint8_t* ws = (uint8_t*)d_ws;

  size_t off = 0;
  _Float16* xh    = (_Float16*)(ws + off); off += (size_t)M * CH * 2;           // 8 MB
  _Float16* wqkvh = (_Float16*)(ws + off); off += (size_t)3 * CH * CH * 2;      // 6 MB
  _Float16* pwh   = (_Float16*)(ws + off); off += (size_t)CH * CH * 2;          // 2 MB
  float*    qkvf  = (float*)   (ws + off); off += (size_t)M * 3 * CH * 4;       // 48 MB
  _Float16* qh    = (_Float16*)(ws + off); off += (size_t)M * CH * 2;           // 8 MB
  _Float16* kh    = (_Float16*)(ws + off); off += (size_t)M * CH * 2;           // 8 MB
  _Float16* vT    = (_Float16*)(ws + off); off += (size_t)M * CH * 2;           // 8 MB
  _Float16* attno = (_Float16*)(ws + off); off += (size_t)M * CH * 2;           // 8 MB

  // 0) fp32 -> fp16 copies of activations / weights
  cvt_f32_to_f16<<<(M * CH + 255) / 256, 256, 0, stream>>>(x, xh, M * CH);
  cvt_f32_to_f16<<<(3 * CH * CH + 255) / 256, 256, 0, stream>>>(qkv_w, wqkvh, 3 * CH * CH);
  cvt_f32_to_f16<<<(CH * CH + 255) / 256, 256, 0, stream>>>(proj_w, pwh, CH * CH);

  // 1) qkv = x @ qkv_w^T + qkv_b    [4096 x 3072]
  wmma_gemm_bias<<<dim3(M / 256, (3 * CH) / 64), 256, 0, stream>>>(
      xh, wqkvh, qkv_b, qkvf, M, 3 * CH, CH);

  // 2) QK layernorm + head split (+ q*scale) + V transpose
  qk_ln_split<<<M, 256, 0, stream>>>(qkvf, qn_w, qn_b, kn_w, kn_b, qh, kh, vT);

  // 3) flash attention -> attno (f16, [B,N,C])
  flash_attn<<<BATCH * HEADS * (SEQ / 128), 256, 0, stream>>>(qh, kh, vT, attno);

  // 4) out = attno @ proj_w^T + proj_b   (f32 output)
  wmma_gemm_bias<<<dim3(M / 256, CH / 64), 256, 0, stream>>>(
      attno, pwh, proj_b, (float*)d_out, M, CH, CH);
}
